// EmbeddingLayer_33165737459873
// MI455X (gfx1250) — matrix operand order repaired and checked by reading, not compiled, not run on hardware
//
#include <hip/hip_runtime.h>

#define EMB 128
#define NDEV 100000
#define KB 8

typedef __attribute__((ext_vector_type(16))) __bf16 v16bf;
typedef __attribute__((ext_vector_type(8)))  float  v8f;

__device__ __forceinline__ float fast_tanh(float x) {
#if __has_builtin(__builtin_amdgcn_tanhf)
    // CDNA5 has a native transcendental V_TANH_F32 (tracked as TRANS32); one op, co-executes with VALU.
    return __builtin_amdgcn_tanhf(x);
#else
    // Fallback: tanh(x) = sign(x) * (1 - 2/(exp2(2*|x|*log2e) + 1)); uses v_exp_f32 + v_rcp_f32.
    float ax = __builtin_fabsf(x);
    float e  = __builtin_amdgcn_exp2f(ax * 2.8853900817779268f);
    float r  = __builtin_amdgcn_rcpf(e + 1.0f);
    float t  = __builtin_fmaf(-2.0f, r, 1.0f);
    return __builtin_copysignf(t, x);
#endif
}

__global__ __launch_bounds__(512, 1)
void fused_embed_kernel(const float* __restrict__ V_pre,
                        const int*   __restrict__ dev_breakers,
                        const int*   __restrict__ breakers_arr,
                        const float* __restrict__ protector_state,
                        const float* __restrict__ breaker_state,
                        const float* __restrict__ W0, const float* __restrict__ b0,
                        const float* __restrict__ w1, const float* __restrict__ b1,
                        const float* __restrict__ w2, const float* __restrict__ b2,
                        const float* __restrict__ W3, const float* __restrict__ b3,
                        const float* __restrict__ w4, const float* __restrict__ b4,
                        const float* __restrict__ W5, const float* __restrict__ b5,
                        const float* __restrict__ conv_w, const float* __restrict__ conv_b,
                        float* __restrict__ out)
{
    // Weights staged as bf16 WMMA A-fragments (W^T tiles): [3 mats][8 ntile][4 ktile][32 lanes][16 elem]
    __shared__ __attribute__((aligned(64))) __bf16 wfrag[3 * 8 * 4 * 32 * 16];  // 96 KB

    const int tid = threadIdx.x;
    {
        const float* Ws[3] = { W0, W3, W5 };
        for (int idx = tid; idx < 3 * 8 * 4 * 32 * 16; idx += 512) {
            int w    = idx >> 14;          // matrix
            int r    = idx & 16383;
            int nt   = r >> 11;            // n tile
            int r2   = r & 2047;
            int kt   = r2 >> 9;            // k tile
            int r3   = r2 & 511;
            int ln   = r3 >> 4;            // lane
            int e    = r3 & 15;            // element within lane
            int i    = ln & 15;
            int hh   = ln >> 4;
            // 16-bit A-matrix element mapping (contiguous, matches B): K = 16*half + e
            int k    = kt * 32 + (hh << 4) + e;
            int n    = nt * 16 + i;
            // A[i][k] = W^T[n][k] = W[k][n]
            wfrag[idx] = (__bf16)Ws[w][k * EMB + n];
        }
    }
    __syncthreads();

    const int wid  = tid >> 5;
    const int lane = tid & 31;
    const int tile = blockIdx.x * 16 + wid;
    if (tile * 16 >= NDEV) return;         // wave-uniform exit (EXEC stays full for WMMA)

    const int m    = lane & 15;
    const int half = lane >> 4;
    const int d    = tile * 16 + m;

    // ---- per-device scalars ----
    int bidx[KB]; float cs[KB]; int nid[KB];
    float S = 0.0f;
    #pragma unroll
    for (int k = 0; k < KB; ++k) {
        bidx[k] = dev_breakers[d * KB + k];
        cs[k]   = breaker_state[bidx[k]];
        S += cs[k];
    }
    #pragma unroll
    for (int k = 0; k < KB; ++k) {
        int e0 = breakers_arr[bidx[k] * 2 + 0];
        int e1 = breakers_arr[bidx[k] * 2 + 1];
        nid[k] = (e0 == d) ? e1 : e0;
    }
    const float p0 = protector_state[d * 3 + 0];
    const float p1 = protector_state[d * 3 + 1];
    const float p2 = protector_state[d * 3 + 2];

    // ---- phase 1: build WMMA B fragments in registers ----
    // B-operand layout: lane (j=m, half) element e = X[m][kt*32 + 16*half + e]
    v16bf fPT[4], fBE[4], fNB[4];

    #pragma unroll
    for (int kt = 0; kt < 4; ++kt) {
        const int cb = kt * 32 + half * 16;
        #pragma unroll
        for (int g = 0; g < 4; ++g) {
            const int c = cb + g * 4;
            float w4v[4], b4v[4], w1v[4], b1v[4], w2v[4], b2v[4];
            {   // vector loads of the (uniform-per-half) weight slices
                float4 t;
                t = *(const float4*)(w4 + c); w4v[0]=t.x; w4v[1]=t.y; w4v[2]=t.z; w4v[3]=t.w;
                t = *(const float4*)(b4 + c); b4v[0]=t.x; b4v[1]=t.y; b4v[2]=t.z; b4v[3]=t.w;
                t = *(const float4*)(w1 + c); w1v[0]=t.x; w1v[1]=t.y; w1v[2]=t.z; w1v[3]=t.w;
                t = *(const float4*)(b1 + c); b1v[0]=t.x; b1v[1]=t.y; b1v[2]=t.z; b1v[3]=t.w;
                t = *(const float4*)(w2 + c); w2v[0]=t.x; w2v[1]=t.y; w2v[2]=t.z; w2v[3]=t.w;
                t = *(const float4*)(b2 + c); b2v[0]=t.x; b2v[1]=t.y; b2v[2]=t.z; b2v[3]=t.w;
            }
            float sbe[4] = {0.f, 0.f, 0.f, 0.f};
            float snb[4] = {0.f, 0.f, 0.f, 0.f};
            #pragma unroll
            for (int k = 0; k < KB; ++k) {
                float4 vp = *(const float4*)(V_pre + (long)nid[k] * EMB + c);
                snb[0] += vp.x; snb[1] += vp.y; snb[2] += vp.z; snb[3] += vp.w;
                #pragma unroll
                for (int q = 0; q < 4; ++q)
                    sbe[q] += fast_tanh(__builtin_fmaf(cs[k], w4v[q], b4v[q]));
            }
            #pragma unroll
            for (int q = 0; q < 4; ++q) {
                float spt = 3.0f * fast_tanh(__builtin_fmaf(S,  w2v[q], b2v[q]))
                          +        fast_tanh(__builtin_fmaf(p0, w1v[q], b1v[q]))
                          +        fast_tanh(__builtin_fmaf(p1, w1v[q], b1v[q]))
                          +        fast_tanh(__builtin_fmaf(p2, w1v[q], b1v[q]));
                const int e = g * 4 + q;
                fPT[kt][e] = (__bf16)spt;
                fBE[kt][e] = (__bf16)sbe[q];
                fNB[kt][e] = (__bf16)snb[q];
            }
        }
    }

    // ---- phase 2: GEMMs via WMMA + fused epilogue ----
    const float cw0 = conv_w[0], cw1 = conv_w[1], cw2 = conv_w[2], cbv = conv_b[0];

    for (int nt = 0; nt < 8; ++nt) {
        v8f a0 = {}, a3 = {}, a5 = {};
        #pragma unroll
        for (int kt = 0; kt < 4; ++kt) {
            const v16bf A0 = *(const v16bf*)&wfrag[(((0 * 8 + nt) * 4 + kt) * 32 + lane) * 16];
            const v16bf A3 = *(const v16bf*)&wfrag[(((1 * 8 + nt) * 4 + kt) * 32 + lane) * 16];
            const v16bf A5 = *(const v16bf*)&wfrag[(((2 * 8 + nt) * 4 + kt) * 32 + lane) * 16];
            a0 = __builtin_amdgcn_wmma_f32_16x16x32_bf16(false, A0, false, fPT[kt], (short)0, a0, false, false);
            a3 = __builtin_amdgcn_wmma_f32_16x16x32_bf16(false, A3, false, fBE[kt], (short)0, a3, false, false);
            a5 = __builtin_amdgcn_wmma_f32_16x16x32_bf16(false, A5, false, fNB[kt], (short)0, a5, false, false);
        }
        // D[i][j]: lane holds column j=m, rows i = n0..n0+7 (contiguous in emb dim)
        const int n0 = nt * 16 + half * 8;
        float o[8];
        #pragma unroll
        for (int r = 0; r < 8; ++r) {
            float pe = fast_tanh(a0[r] + b0[n0 + r]);
            float be = fast_tanh(a3[r] + b3[n0 + r]);
            float ne = fast_tanh(a5[r] + b5[n0 + r]);
            o[r] = fast_tanh(__builtin_fmaf(cw0, pe,
                     __builtin_fmaf(cw1, be,
                     __builtin_fmaf(cw2, ne, cbv))));
        }
        float4* op = (float4*)(out + (long)d * EMB + n0);
        op[0] = make_float4(o[0], o[1], o[2], o[3]);
        op[1] = make_float4(o[4], o[5], o[6], o[7]);
    }
}

extern "C" void kernel_launch(void* const* d_in, const int* in_sizes, int n_in,
                              void* d_out, int out_size, void* d_ws, size_t ws_size,
                              hipStream_t stream) {
    (void)in_sizes; (void)n_in; (void)out_size; (void)d_ws; (void)ws_size;
    const float* V_pre           = (const float*)d_in[0];
    const int*   dev_breakers    = (const int*)  d_in[1];
    const int*   breakers_arr    = (const int*)  d_in[2];
    const float* protector_state = (const float*)d_in[3];
    const float* breaker_state   = (const float*)d_in[4];
    const float* W0 = (const float*)d_in[5];  const float* b0 = (const float*)d_in[6];
    const float* w1 = (const float*)d_in[7];  const float* b1 = (const float*)d_in[8];
    const float* w2 = (const float*)d_in[9];  const float* b2 = (const float*)d_in[10];
    const float* W3 = (const float*)d_in[11]; const float* b3 = (const float*)d_in[12];
    const float* w4 = (const float*)d_in[13]; const float* b4 = (const float*)d_in[14];
    const float* W5 = (const float*)d_in[15]; const float* b5 = (const float*)d_in[16];
    const float* conv_w = (const float*)d_in[17];
    const float* conv_b = (const float*)d_in[18];

    const int tiles  = (NDEV + 15) / 16;       // 6250 tiles of 16 devices
    const int blocks = (tiles + 15) / 16;      // 16 waves (tiles) per 512-thread block

    fused_embed_kernel<<<blocks, 512, 0, stream>>>(
        V_pre, dev_breakers, breakers_arr, protector_state, breaker_state,
        W0, b0, w1, b1, w2, b2, W3, b3, w4, b4, W5, b5, conv_w, conv_b,
        (float*)d_out);
}
